// HDSGNN_90744069030457
// MI455X (gfx1250) — compile-verified, matrix operand order checked by reading, not run on hardware
//
#include <hip/hip_runtime.h>
#include <hip/hip_bf16.h>

// ---------------------------------------------------------------------------
// HDSGNN forward on MI455X (gfx1250), wave32 + WMMA bf16 + async-LDS staging.
// ---------------------------------------------------------------------------

#define N_NODES 50000
#define FEAT    128
#define N_EDGES 1600000
#define NHID    256
#define NGRP    512
#define NCLS    10

typedef __attribute__((ext_vector_type(16))) __bf16 v16bf;
typedef __attribute__((ext_vector_type(8)))  float  v8f;

#if defined(__AMDGCN__) && __has_builtin(__builtin_amdgcn_global_load_async_to_lds_b128) && __has_builtin(__builtin_amdgcn_s_wait_asynccnt)
#define USE_ASYNC_LDS 1
typedef int v4i __attribute__((ext_vector_type(4)));
typedef __attribute__((address_space(1))) v4i as1_v4i;   // global b128 chunk
typedef __attribute__((address_space(3))) v4i as3_v4i;   // LDS b128 chunk
#else
#define USE_ASYNC_LDS 0
#endif

// ---------------------------------------------------------------------------
// degree / normalization
// ---------------------------------------------------------------------------
__global__ void init_deg_kernel(float* __restrict__ deg, int n) {
    int i = blockIdx.x * blockDim.x + threadIdx.x;
    if (i < n) deg[i] = 1.0f;   // self loop
}

__global__ void edge_deg_kernel(const int* __restrict__ dst, float* __restrict__ deg, int e) {
    int i = blockIdx.x * blockDim.x + threadIdx.x;
    if (i < e)
        __hip_atomic_fetch_add(&deg[dst[i]], 1.0f, __ATOMIC_RELAXED, __HIP_MEMORY_SCOPE_AGENT);
}

__global__ void rsqrt_kernel(float* __restrict__ deg, int n) {
    int i = blockIdx.x * blockDim.x + threadIdx.x;
    if (i < n) deg[i] = rsqrtf(deg[i]);
}

// ---------------------------------------------------------------------------
// Weight pre-pack: W[K,256] f32 -> bf16 fragments in per-lane WMMA order.
// Element layout: Wp[(((kblk*16 + t)*32 + lane)*16 + j)]
//   lane half kb = (lane>>4)*8, col = t*16 + (lane&15)
//   j = 2v(+1): K index = kblk*32 + kb + (v<4 ? 2v : 16+2(v-4)) (+1)
// One thread produces one 16-element (32-byte) fragment.
// ---------------------------------------------------------------------------
__global__ void pack_weights_kernel(const float* __restrict__ W, __bf16* __restrict__ Wp, int K)
{
    int idx = blockIdx.x * blockDim.x + threadIdx.x;    // over (K/32)*16*32
    int total = (K >> 5) * 16 * 32;
    if (idx >= total) return;
    int lane = idx & 31;
    int t    = (idx >> 5) & 15;
    int kblk = idx >> 9;
    int k0   = kblk * 32;
    int kb   = (lane >> 4) * 8;
    int col  = t * 16 + (lane & 15);
    v16bf frag;
    #pragma unroll
    for (int v = 0; v < 8; ++v) {
        int kk = kb + ((v < 4) ? (2 * v) : (16 + 2 * (v - 4)));
        frag[2 * v]     = (__bf16)W[(long)(k0 + kk)     * NHID + col];
        frag[2 * v + 1] = (__bf16)W[(long)(k0 + kk + 1) * NHID + col];
    }
    *(v16bf*)(Wp + (size_t)idx * 16) = frag;
}

// ---------------------------------------------------------------------------
// WMMA bf16 GEMM:  C[n, :256] = act(A[n, :K] @ W[K,256] + bias)
// Block = 128 threads (4 waves); block covers 64 rows, wave w covers rows
// [blk*64 + 16w, +16) x all 256 cols = 16 accumulator tiles of 16x16.
// Packed B for one k-step (16KB, contiguous) is double-buffered through LDS
// with GLOBAL_LOAD_ASYNC_TO_LDS_B128; B-fragment reads are software-pipelined
// one tile ahead so ds latency overlaps the matrix pipe.
// DIFF mode: A[n,k] = diff[(k>>7)*chunkStride + n*128 + (k&127)] * scale(k>>7)
// ---------------------------------------------------------------------------
template<bool DIFF, bool RELU, bool BIAS>
__global__ __launch_bounds__(128)
void gemm_wmma_kernel(const float* __restrict__ A, int lda, long chunkStride,
                      const float* __restrict__ scales, int K,
                      const __bf16* __restrict__ Wp,
                      const float* __restrict__ bias,
                      float* __restrict__ C, int ldc, int coff)
{
    const int tid  = threadIdx.x;
    const int lane = tid & 31;
    const int wave = tid >> 5;
    const int m0   = blockIdx.x * 64 + wave * 16;
    const int lm   = lane & 15;          // 0..15
    const int kb   = (lane >> 4) * 8;    // 0 or 8 (half-wave K base)
    int row = m0 + lm;                   // A row this lane supplies
    if (row >= N_NODES) row = N_NODES - 1;

    const int nK = K >> 5;               // number of 32-wide k-steps

    v8f acc[16];
    #pragma unroll
    for (int t = 0; t < 16; ++t) acc[t] = v8f{};

#if USE_ASYNC_LDS
    __shared__ __bf16 smemB[2][8192];    // 2 x 16KB double buffer
    // issue batch for k-step 0 into buffer 0 (8 x b128 per thread)
    {
        as1_v4i* g = (as1_v4i*)(uintptr_t)(const void*)(Wp + (size_t)tid * 64);
        as3_v4i* l = (as3_v4i*)(uint32_t)(uintptr_t)(void*)(&smemB[0][tid * 64]);
        #pragma unroll
        for (int j = 0; j < 8; ++j)
            __builtin_amdgcn_global_load_async_to_lds_b128(g + j, l + j, 0, 0);
    }
#endif

    for (int kblk = 0; kblk < nK; ++kblk) {
        const int k0 = kblk * 32;

#if USE_ASYNC_LDS
        const int buf = kblk & 1;
        if (kblk + 1 < nK) {
            // prefetch next k-step into the other buffer
            as1_v4i* g = (as1_v4i*)(uintptr_t)(const void*)(Wp + (size_t)(kblk + 1) * 8192 + (size_t)tid * 64);
            as3_v4i* l = (as3_v4i*)(uint32_t)(uintptr_t)(void*)(&smemB[buf ^ 1][tid * 64]);
            #pragma unroll
            for (int j = 0; j < 8; ++j)
                __builtin_amdgcn_global_load_async_to_lds_b128(g + j, l + j, 0, 0);
            __builtin_amdgcn_s_wait_asynccnt(8);   // batch kblk done, next in flight
        } else {
            __builtin_amdgcn_s_wait_asynccnt(0);
        }
        __syncthreads();
#endif

        // ---- build A fragment (16x32 bf16) ----
        const float* Arow;
        float sc = 1.0f;
        if (DIFF) {
            int c = k0 >> 7;                      // 128-wide chunk, constant per k-step
            Arow = A + (long)c * chunkStride + (long)row * FEAT + (k0 & 127);
            if (scales) sc = scales[c % 3];
        } else {
            Arow = A + (long)row * lda + k0;
        }
        v16bf a;
        #pragma unroll
        for (int v = 0; v < 8; ++v) {
            int kk = kb + ((v < 4) ? (2 * v) : (16 + 2 * (v - 4)));
            float2 av = *(const float2*)(Arow + kk);
            a[2 * v]     = (__bf16)(av.x * sc);
            a[2 * v + 1] = (__bf16)(av.y * sc);
        }

        // ---- 16 col-tiles: B fragments pipelined one tile ahead ----
#if USE_ASYNC_LDS
        #define B_FRAG(t) (*(const v16bf*)&smemB[buf][((t) * 32 + lane) * 16])
#else
        #define B_FRAG(t) (*(const v16bf*)(Wp + ((size_t)kblk * 16 + (t)) * 512 + (size_t)lane * 16))
#endif
        v16bf bcur = B_FRAG(0);
        #pragma unroll
        for (int t = 0; t < 16; ++t) {
            v16bf bnxt = bcur;
            if (t < 15) bnxt = B_FRAG(t + 1);   // in flight during wmma t
            acc[t] = __builtin_amdgcn_wmma_f32_16x16x32_bf16(
                false, a, false, bcur, (short)0, acc[t], false, false);
            bcur = bnxt;
        }
        #undef B_FRAG

#if USE_ASYNC_LDS
        __syncthreads();   // all waves done reading buf before it is refilled
#endif
    }

    // ---- store C (16x16 f32 tile: VGPR r -> M = r + (lane>=16 ? 8:0)) ----
    const int rbase = m0 + ((lane >> 4) ? 8 : 0);
    #pragma unroll
    for (int t = 0; t < 16; ++t) {
        const int cc = t * 16 + lm;
        float bv = BIAS ? bias[cc] : 0.0f;
        #pragma unroll
        for (int r = 0; r < 8; ++r) {
            int rr = rbase + r;
            if (rr < N_NODES) {
                float v = acc[t][r] + bv;
                if (RELU) v = fmaxf(v, 0.0f);
                C[(long)rr * ldc + coff + cc] = v;
            }
        }
    }
}

// ---------------------------------------------------------------------------
// GCN aggregation: out[n] = bias + dinv[n]^2 * h[n]  (self loop + bias init)
// ---------------------------------------------------------------------------
__global__ void conv_self_bias_kernel(const float* __restrict__ h,
                                      const float* __restrict__ dinv,
                                      const float* __restrict__ bias,
                                      float* __restrict__ out, int ldo, int off)
{
    long idx = (long)blockIdx.x * 256 + threadIdx.x;     // over N*256
    int n = (int)(idx >> 8), c = (int)(idx & 255);
    if (n >= N_NODES) return;
    float dv = dinv[n];
    out[(long)n * ldo + off + c] = h[(long)n * NHID + c] * dv * dv + bias[c];
}

// one wave per edge; lane owns 8 consecutive features (256 = 32 lanes * 8)
__global__ __launch_bounds__(256)
void scatter_edges_kernel(const int* __restrict__ src, const int* __restrict__ dst,
                          const float* __restrict__ h, const float* __restrict__ dinv,
                          float* __restrict__ out, int ldo, int off)
{
    int e = blockIdx.x * (blockDim.x >> 5) + (threadIdx.x >> 5);
    if (e >= N_EDGES) return;
    int lane = threadIdx.x & 31;
    int s = src[e], d = dst[e];
    float nrm = dinv[s] * dinv[d];
    const float4* hs = (const float4*)(h + (long)s * NHID);
    float* od = out + (long)d * ldo + off;
    #pragma unroll
    for (int j = 0; j < 2; ++j) {
        float4 v = hs[lane * 2 + j];
        int b = lane * 8 + j * 4;
        __hip_atomic_fetch_add(od + b + 0, v.x * nrm, __ATOMIC_RELAXED, __HIP_MEMORY_SCOPE_AGENT);
        __hip_atomic_fetch_add(od + b + 1, v.y * nrm, __ATOMIC_RELAXED, __HIP_MEMORY_SCOPE_AGENT);
        __hip_atomic_fetch_add(od + b + 2, v.z * nrm, __ATOMIC_RELAXED, __HIP_MEMORY_SCOPE_AGENT);
        __hip_atomic_fetch_add(od + b + 3, v.w * nrm, __ATOMIC_RELAXED, __HIP_MEMORY_SCOPE_AGENT);
    }
}

__global__ void relu_kernel(float* __restrict__ out, int ldo, int off)
{
    long idx = (long)blockIdx.x * 256 + threadIdx.x;
    int n = (int)(idx >> 8), c = (int)(idx & 255);
    if (n >= N_NODES) return;
    float* p = out + (long)n * ldo + off + c;
    *p = fmaxf(*p, 0.0f);
}

// ---------------------------------------------------------------------------
// pooling + final FC
// ---------------------------------------------------------------------------
__global__ void zero_kernel(float* __restrict__ p, int n) {
    int i = blockIdx.x * blockDim.x + threadIdx.x;
    if (i < n) p[i] = 0.0f;
}

__global__ void pool_kernel(const float* __restrict__ clsx, const int* __restrict__ batch,
                            float* __restrict__ pooled)
{
    long idx = (long)blockIdx.x * 256 + threadIdx.x;
    int n = (int)(idx >> 8), c = (int)(idx & 255);
    if (n >= N_NODES) return;
    __hip_atomic_fetch_add(&pooled[(long)batch[n] * NHID + c],
                           clsx[(long)n * NHID + c],
                           __ATOMIC_RELAXED, __HIP_MEMORY_SCOPE_AGENT);
}

__global__ void final_fc_kernel(const float* __restrict__ pooled,
                                const float* __restrict__ Wfc,
                                const float* __restrict__ bfc,
                                float* __restrict__ out)
{
    int idx = blockIdx.x * blockDim.x + threadIdx.x;
    if (idx >= NGRP * NCLS) return;
    int g = idx / NCLS, c = idx % NCLS;
    float acc = bfc[c];
    const float* pg = pooled + (long)g * NHID;
    #pragma unroll 8
    for (int k = 0; k < NHID; ++k) acc += pg[k] * Wfc[k * NCLS + c];
    out[idx] = acc;
}

// ---------------------------------------------------------------------------
// host launcher
// ---------------------------------------------------------------------------
extern "C" void kernel_launch(void* const* d_in, const int* in_sizes, int n_in,
                              void* d_out, int out_size, void* d_ws, size_t ws_size,
                              hipStream_t stream)
{
    // inputs in setup_inputs() dict order
    const float* x        = (const float*)d_in[0];
    const int*   eidx     = (const int*)d_in[1];
    const float* diff     = (const float*)d_in[2];
    const int*   batch    = (const int*)d_in[3];
    const float* Wc[3]    = { (const float*)d_in[4],  (const float*)d_in[8],  (const float*)d_in[12] };
    const float* bc[3]    = { (const float*)d_in[5],  (const float*)d_in[9],  (const float*)d_in[13] };
    const float* Wl[3]    = { (const float*)d_in[6],  (const float*)d_in[10], (const float*)d_in[14] };
    const float* bl[3]    = { (const float*)d_in[7],  (const float*)d_in[11], (const float*)d_in[15] };
    const float* ow       = (const float*)d_in[16];
    const float* Wcls     = (const float*)d_in[17];
    const float* bcls     = (const float*)d_in[18];
    const float* Wfc      = (const float*)d_in[19];
    const float* bfc      = (const float*)d_in[20];
    float* out            = (float*)d_out;

    const int* esrc = eidx;
    const int* edst = eidx + N_EDGES;

    // workspace carve
    float*  dinv   = (float*)d_ws;                         // N (rounded up)
    float*  hW     = dinv + 50432;                         // N*256
    float*  comb   = hW + (size_t)N_NODES * NHID;          // N*512
    float*  pooled = comb + (size_t)N_NODES * 2 * NHID;    // G*256
    __bf16* wpack  = (__bf16*)(pooled + NGRP * NHID);      // packed weights (bf16)

    // packed weight offsets (elements): order c0,l0,c1,l1,c2,l2,cls
    const int convK[3] = { FEAT, 2 * NHID, 2 * NHID };
    const int linK[3]  = { FEAT, 3 * FEAT, 9 * FEAT };
    __bf16* wpC[3]; __bf16* wpL[3]; __bf16* wpCls;
    {
        size_t off = 0;
        for (int i = 0; i < 3; ++i) {
            wpC[i] = wpack + off; off += (size_t)convK[i] * NHID;
            wpL[i] = wpack + off; off += (size_t)linK[i]  * NHID;
        }
        wpCls = wpack + off;
    }

    const int TPB = 256;
    const dim3 gN((N_NODES + TPB - 1) / TPB);
    const dim3 gE((N_EDGES + TPB - 1) / TPB);
    const dim3 gNC(N_NODES);                    // N*256 element kernels, 256/blk
    const dim3 gEdgeWaves((N_EDGES + 7) / 8);   // 8 waves per block
    const dim3 gGemm((N_NODES + 63) / 64);      // 64 rows per block
    const long chunkStride = (long)N_NODES * FEAT;

    // ---- pack all weights into WMMA fragment order (bf16) ----
    for (int i = 0; i < 3; ++i) {
        pack_weights_kernel<<<(convK[i] * 16 + TPB - 1) / TPB, TPB, 0, stream>>>(Wc[i], wpC[i], convK[i]);
        pack_weights_kernel<<<(linK[i]  * 16 + TPB - 1) / TPB, TPB, 0, stream>>>(Wl[i], wpL[i], linK[i]);
    }
    pack_weights_kernel<<<(2 * NHID * 16 + TPB - 1) / TPB, TPB, 0, stream>>>(Wcls, wpCls, 2 * NHID);

    // ---- degree / norm ----
    init_deg_kernel<<<gN, TPB, 0, stream>>>(dinv, N_NODES);
    edge_deg_kernel<<<gE, TPB, 0, stream>>>(edst, dinv, N_EDGES);
    rsqrt_kernel<<<gN, TPB, 0, stream>>>(dinv, N_NODES);

    const long diffOff[3] = { 0, 1 * chunkStride, 4 * chunkStride };

    for (int i = 0; i < 3; ++i) {
        const float* Ain  = (i == 0) ? x : comb;
        const int    ldin = (i == 0) ? FEAT : 2 * NHID;
        // conv GEMM: (prev comb) @ Wc -> hW  (no bias/relu yet)
        gemm_wmma_kernel<false, false, false><<<gGemm, 128, 0, stream>>>(
            Ain, ldin, 0, nullptr, convK[i], wpC[i], nullptr, hW, NHID, 0);
        // self-loop + bias into right half of comb
        conv_self_bias_kernel<<<gNC, TPB, 0, stream>>>(hW, dinv, bc[i], comb, 2 * NHID, NHID);
        // edge scatter-add
        scatter_edges_kernel<<<gEdgeWaves, TPB, 0, stream>>>(esrc, edst, hW, dinv,
                                                             comb, 2 * NHID, NHID);
        // relu on conv half
        relu_kernel<<<gNC, TPB, 0, stream>>>(comb, 2 * NHID, NHID);
        // lin GEMM from diff_feat chunks (scaled for i>0), bias+relu fused,
        // into left half of comb
        gemm_wmma_kernel<true, true, true><<<gGemm, 128, 0, stream>>>(
            diff + diffOff[i], 0, chunkStride, (i > 0) ? ow : nullptr,
            linK[i], wpL[i], bl[i], comb, 2 * NHID, 0);
    }

    // ---- classifier GCN (no relu) ----
    gemm_wmma_kernel<false, false, false><<<gGemm, 128, 0, stream>>>(
        comb, 2 * NHID, 0, nullptr, 2 * NHID, wpCls, nullptr, hW, NHID, 0);
    // reuse comb as N x 256 accumulation buffer for the classifier output
    conv_self_bias_kernel<<<gNC, TPB, 0, stream>>>(hW, dinv, bcls, comb, NHID, 0);
    scatter_edges_kernel<<<gEdgeWaves, TPB, 0, stream>>>(esrc, edst, hW, dinv, comb, NHID, 0);

    // ---- pool + FC ----
    zero_kernel<<<(NGRP * NHID + TPB - 1) / TPB, TPB, 0, stream>>>(pooled, NGRP * NHID);
    pool_kernel<<<gNC, TPB, 0, stream>>>(comb, batch, pooled);
    final_fc_kernel<<<(NGRP * NCLS + TPB - 1) / TPB, TPB, 0, stream>>>(pooled, Wfc, bfc, out);
}
